// GE_PINN_39994735460583
// MI455X (gfx1250) — matrix operand order repaired and checked by reading, not compiled, not run on hardware
//
#include <hip/hip_runtime.h>
#include <hip/hip_fp16.h>

typedef __attribute__((ext_vector_type(16))) _Float16 v16h;
typedef __attribute__((ext_vector_type(8)))  _Float16 v8h;
typedef __attribute__((ext_vector_type(8)))  float    v8f;

#define CCH 16
#define NBF 10

__device__ __forceinline__ float silu_f(float x)    { return x / (1.0f + __expf(-x)); }
__device__ __forceinline__ float sigmoid_f(float x) { return 1.0f / (1.0f + __expf(-x)); }

// ---------------------------------------------------------------------------
// Pack a dense [K x Ncols] f32 weight into WMMA B-fragment order (f16):
// frag[((kc*nT + t)*32 + lane)*16 + e]  holds  W[kc*32 + (lane&16?8:0) + (e&7) + (e&8?16:0)][t*16 + (lane&15)]
// zero-padded outside [K, Ncols]; scale folded in.
// ---------------------------------------------------------------------------
__global__ void pack_b_kernel(const float* __restrict__ W, _Float16* __restrict__ frag,
                              int K, int Ncols, int nKC, int nT, float scale)
{
    int tid = blockIdx.x * blockDim.x + threadIdx.x;
    int total = nKC * nT * 32 * 16;
    if (tid >= total) return;
    int e    = tid & 15;
    int lane = (tid >> 4) & 31;
    int t    = (tid >> 9) % nT;
    int kc   = (tid >> 9) / nT;
    int k = kc * 32 + ((lane & 16) ? 8 : 0) + (e & 7) + ((e & 8) ? 16 : 0);
    int n = t * 16 + (lane & 15);
    float v = (k < K && n < Ncols) ? W[k * Ncols + n] * scale : 0.0f;
    frag[tid] = (_Float16)v;
}

// ---------------------------------------------------------------------------
// Edge geometry: Y1 (component-normalized l=1 SH, z==0) and edge length.
// ---------------------------------------------------------------------------
__global__ void geom_kernel(const float* __restrict__ x, const int* __restrict__ src,
                            const int* __restrict__ dst, float* __restrict__ Y1,
                            float* __restrict__ elen, int E)
{
    int e = blockIdx.x * blockDim.x + threadIdx.x;
    if (e >= E) return;
    int si = src[e], di = dst[e];
    float ax = x[si * 2 + 0] - 1.0f;
    float ay = (x[si * 2 + 1] - 0.5f) * 2.0f;
    float bx = x[di * 2 + 0] - 1.0f;
    float by = (x[di * 2 + 1] - 0.5f) * 2.0f;
    float rx = ax - bx, ry = ay - by;
    float len = sqrtf(rx * rx + ry * ry + 1e-12f);
    float s = 1.7320508f / len; // sqrt(3)
    Y1[(size_t)e * 3 + 0] = rx * s;
    Y1[(size_t)e * 3 + 1] = ry * s;
    Y1[(size_t)e * 3 + 2] = 0.0f;
    elen[e] = len;
}

__global__ void zero_kernel(float* __restrict__ p, long total)
{
    long i = (long)blockIdx.x * blockDim.x + threadIdx.x;
    if (i < total) p[i] = 0.0f;
}

__global__ void fill_s0_kernel(float* __restrict__ s, const float* __restrict__ embed, int total)
{
    int i = blockIdx.x * blockDim.x + threadIdx.x;
    if (i < total) s[i] = embed[i & 15];
}

// ---------------------------------------------------------------------------
// Per-layer edge kernel: radial MLP (two WMMA GEMMs) fused with TP message
// construction and atomic scatter-add into agg[N][80] = [sf(16) sg(16) vm(48)].
// One wave = 16 edges (WMMA M); 8 waves per block.
// ---------------------------------------------------------------------------
__global__ __launch_bounds__(256)
void edge_conv_kernel(const float* __restrict__ elen, const float* __restrict__ Y1,
                      const int* __restrict__ src, const int* __restrict__ dst,
                      const float* __restrict__ s_in, const float* __restrict__ v_in,
                      const _Float16* __restrict__ fragB1, const _Float16* __restrict__ fragB2,
                      float* __restrict__ agg, int E)
{
    __shared__ __align__(16) _Float16 h_lds[8][16 * 112]; // 28 KB: hidden acts (f16)
    __shared__ __align__(16) float    w_lds[8][16 * 96];  // 48 KB: TP weights (f32)

    const int lane = threadIdx.x & 31;
    const int wv   = threadIdx.x >> 5;
    const int e0   = (blockIdx.x * 8 + wv) * 16;
    const int m    = lane & 15;
    const int half = lane >> 4;
    const int koff = half * 8;

    // prefetch next tile's streaming edge data (global_prefetch_b8)
    if (e0 + 4096 < E) {
        __builtin_prefetch(elen + e0 + 4096, 0, 3);
        __builtin_prefetch(Y1 + (size_t)(e0 + 4096) * 3, 0, 3);
    }

    // ---- A1 fragment: gaussian radial basis, built directly in registers ----
    int erow = e0 + m; if (erow > E - 1) erow = E - 1;
    float dn = elen[erow] * (9.0f / 0.3f);     // d / step,  step = R_MAX/(NB-1)
    const float BSCALE = 2.8234622f;           // sqrt(10)/1.12
    v16h a1;
    #pragma unroll
    for (int j = 0; j < 8; ++j) {
        int k = koff + j;                      // elements 0..7 -> K = koff+j
        float diff = dn - (float)k;
        float val = (k < NBF) ? __expf(-diff * diff) * BSCALE : 0.0f;
        a1[j]     = (_Float16)val;
        a1[8 + j] = (_Float16)0.0f;            // elements 8..15 -> K >= 16 > NBF-1
    }

    // ---- GEMM1: [16x32] x [32x112] -> silu -> h (f16) in LDS ----
    #pragma unroll
    for (int t = 0; t < 7; ++t) {
        v16h b = *(const v16h*)(fragB1 + (t * 32 + lane) * 16);
        v8f c = {};
        c = __builtin_amdgcn_wmma_f32_16x16x32_f16(false, a1, false, b, (short)0, c, false, false);
        int col = t * 16 + m;
        int rb  = half * 8;
        #pragma unroll
        for (int r = 0; r < 8; ++r)
            h_lds[wv][(rb + r) * 112 + col] = (_Float16)silu_f(c[r]);
    }
    __syncthreads();

    // ---- Rebuild h as A fragments (two aligned ds_load_b128 per K-chunk) ----
    v16h a2[4];
    {
        const _Float16* row = &h_lds[wv][m * 112];
        #pragma unroll
        for (int kc = 0; kc < 4; ++kc) {
            v8h lo = *(const v8h*)(row + kc * 32 + koff);
            v8h hi = *(const v8h*)(row + kc * 32 + 16 + koff);
            #pragma unroll
            for (int j = 0; j < 8; ++j) { a2[kc][j] = lo[j]; a2[kc][8 + j] = hi[j]; }
        }
    }
    __syncthreads();

    // ---- GEMM2: [16x128] x [128x96] -> per-edge TP weights (f32) in LDS ----
    #pragma unroll
    for (int t = 0; t < 6; ++t) {
        v8f c = {};
        #pragma unroll
        for (int kc = 0; kc < 4; ++kc) {
            v16h b = *(const v16h*)(fragB2 + ((kc * 6 + t) * 32 + lane) * 16);
            c = __builtin_amdgcn_wmma_f32_16x16x32_f16(false, a2[kc], false, b, (short)0, c, false, false);
        }
        int col = t * 16 + m;
        int rb  = half * 8;
        #pragma unroll
        for (int r = 0; r < 8; ++r)
            w_lds[wv][(rb + r) * 96 + col] = c[r];
    }
    __syncthreads();

    // ---- Message construction + scatter-add (2 edges x 16 channels / iter) ----
    const float R2N = 0.15811388f; // (1/sqrt(2)) * (1/sqrt(NUM_NEIGH))
    #pragma unroll
    for (int it = 0; it < 8; ++it) {
        int idx = it * 32 + lane;
        int el  = idx >> 4;
        int ch  = idx & 15;
        int e   = e0 + el;
        if (e < E) {
            int sI = src[e], dI = dst[e];
            float ss = s_in[(size_t)sI * CCH + ch];
            const float* vp = v_in + (size_t)sI * 48 + ch * 3;
            float vx = vp[0], vy = vp[1], vz = vp[2];
            const float* yp = Y1 + (size_t)e * 3;
            float y0 = yp[0], y1 = yp[1], y2 = yp[2];
            float dot = vx * y0 + vy * y1 + vz * y2;
            const float* wr = &w_lds[wv][el * 96];
            float w1a = wr[ch],      w1b = wr[16 + ch], w2c = wr[32 + ch];
            float w3c = wr[48 + ch], w4a = wr[64 + ch], w4b = wr[80 + ch];
            float sf = (w1a * ss + w4a * dot) * R2N;
            float sg = (w1b * ss + w4b * dot) * R2N;
            float t0 = w2c * ss;
            float vmx = (t0 * y0 + w3c * vx) * R2N;
            float vmy = (t0 * y1 + w3c * vy) * R2N;
            float vmz = (t0 * y2 + w3c * vz) * R2N;
            float* ag = agg + (size_t)dI * 80;
            atomicAdd(ag + ch,              sf);
            atomicAdd(ag + 16 + ch,         sg);
            atomicAdd(ag + 32 + ch * 3 + 0, vmx);
            atomicAdd(ag + 32 + ch * 3 + 1, vmy);
            atomicAdd(ag + 32 + ch * 3 + 2, vmz);
        }
    }
}

// ---------------------------------------------------------------------------
// Final edge kernel: radial MLP -> [E,64] weights; channel-reduced messages
// (p, vx, vy, vz) via wave shuffles; 4 atomics per edge into aggF[N][4].
// ---------------------------------------------------------------------------
__global__ __launch_bounds__(256)
void edge_final_kernel(const float* __restrict__ elen, const float* __restrict__ Y1,
                       const int* __restrict__ src, const int* __restrict__ dst,
                       const float* __restrict__ s_in, const float* __restrict__ v_in,
                       const _Float16* __restrict__ fragB1, const _Float16* __restrict__ fragB2,
                       float* __restrict__ aggF, int E)
{
    __shared__ __align__(16) _Float16 h_lds[8][16 * 112]; // 28 KB
    __shared__ __align__(16) float    w_lds[8][16 * 64];  // 32 KB

    const int lane = threadIdx.x & 31;
    const int wv   = threadIdx.x >> 5;
    const int e0   = (blockIdx.x * 8 + wv) * 16;
    const int m    = lane & 15;
    const int half = lane >> 4;
    const int koff = half * 8;

    int erow = e0 + m; if (erow > E - 1) erow = E - 1;
    float dn = elen[erow] * (9.0f / 0.3f);
    const float BSCALE = 2.8234622f;
    v16h a1;
    #pragma unroll
    for (int j = 0; j < 8; ++j) {
        int k = koff + j;
        float diff = dn - (float)k;
        float val = (k < NBF) ? __expf(-diff * diff) * BSCALE : 0.0f;
        a1[j] = (_Float16)val;
        a1[8 + j] = (_Float16)0.0f;
    }

    #pragma unroll
    for (int t = 0; t < 7; ++t) {
        v16h b = *(const v16h*)(fragB1 + (t * 32 + lane) * 16);
        v8f c = {};
        c = __builtin_amdgcn_wmma_f32_16x16x32_f16(false, a1, false, b, (short)0, c, false, false);
        int col = t * 16 + m;
        int rb  = half * 8;
        #pragma unroll
        for (int r = 0; r < 8; ++r)
            h_lds[wv][(rb + r) * 112 + col] = (_Float16)silu_f(c[r]);
    }
    __syncthreads();

    v16h a2[4];
    {
        const _Float16* row = &h_lds[wv][m * 112];
        #pragma unroll
        for (int kc = 0; kc < 4; ++kc) {
            v8h lo = *(const v8h*)(row + kc * 32 + koff);
            v8h hi = *(const v8h*)(row + kc * 32 + 16 + koff);
            #pragma unroll
            for (int j = 0; j < 8; ++j) { a2[kc][j] = lo[j]; a2[kc][8 + j] = hi[j]; }
        }
    }
    __syncthreads();

    #pragma unroll
    for (int t = 0; t < 4; ++t) {
        v8f c = {};
        #pragma unroll
        for (int kc = 0; kc < 4; ++kc) {
            v16h b = *(const v16h*)(fragB2 + ((kc * 4 + t) * 32 + lane) * 16);
            c = __builtin_amdgcn_wmma_f32_16x16x32_f16(false, a2[kc], false, b, (short)0, c, false, false);
        }
        int col = t * 16 + m;
        int rb  = half * 8;
        #pragma unroll
        for (int r = 0; r < 8; ++r)
            w_lds[wv][(rb + r) * 64 + col] = c[r];
    }
    __syncthreads();

    const float CNN = 0.039528471f; // (1/sqrt(2C)) * (1/sqrt(NUM_NEIGH))
    #pragma unroll
    for (int it = 0; it < 8; ++it) {
        int idx = it * 32 + lane;
        int el  = idx >> 4;
        int ch  = idx & 15;
        int e   = e0 + el;
        int ec  = (e < E) ? e : (E - 1);
        int sI = src[ec], dI = dst[ec];
        float ss = s_in[(size_t)sI * CCH + ch];
        const float* vp = v_in + (size_t)sI * 48 + ch * 3;
        float vx = vp[0], vy = vp[1], vz = vp[2];
        const float* yp = Y1 + (size_t)ec * 3;
        float y0 = yp[0], y1 = yp[1], y2 = yp[2];
        float dot = vx * y0 + vy * y1 + vz * y2;
        const float* wr = &w_lds[wv][el * 64];
        float wp1 = wr[ch], wp2 = wr[16 + ch], wv1 = wr[32 + ch], wv2 = wr[48 + ch];
        float pp  = (wp1 * ss + wp2 * dot) * CNN;
        float pvx = (wv1 * ss * y0 + wv2 * vx) * CNN;
        float pvy = (wv1 * ss * y1 + wv2 * vy) * CNN;
        float pvz = (wv1 * ss * y2 + wv2 * vz) * CNN;
        #pragma unroll
        for (int off = 8; off > 0; off >>= 1) { // reduce over 16 channel lanes
            pp  += __shfl_xor(pp,  off, 32);
            pvx += __shfl_xor(pvx, off, 32);
            pvy += __shfl_xor(pvy, off, 32);
            pvz += __shfl_xor(pvz, off, 32);
        }
        if (ch == 0 && e < E) {
            float* ag = aggF + (size_t)dI * 4;
            atomicAdd(ag + 0, pp);
            atomicAdd(ag + 1, pvx);
            atomicAdd(ag + 2, pvy);
            atomicAdd(ag + 3, pvz);
        }
    }
}

// ---------------------------------------------------------------------------
// Node scalar update: s_out = s@Wsc_s/4 + agg_s@Wlin_s/sqrt(32); silu/sigmoid.
// ---------------------------------------------------------------------------
__global__ void node_s_kernel(const float* __restrict__ s_in, const float* __restrict__ agg,
                              const float* __restrict__ Wsc_s, const float* __restrict__ Wlin_s,
                              float* __restrict__ s_out, float* __restrict__ gate, int N)
{
    int tid = blockIdx.x * blockDim.x + threadIdx.x;
    if (tid >= N * 32) return;
    int n = tid >> 5, j = tid & 31;
    const float* sr = s_in + (size_t)n * 16;
    const float* ar = agg + (size_t)n * 80;
    float a = 0.0f, b = 0.0f;
    #pragma unroll
    for (int c = 0; c < 16; ++c) a += sr[c] * Wsc_s[c * 32 + j];
    #pragma unroll
    for (int k = 0; k < 32; ++k) b += ar[k] * Wlin_s[k * 32 + j];
    float r = a * 0.25f + b * 0.17677670f;
    if (j < 16) s_out[(size_t)n * 16 + j] = silu_f(r);
    else        gate[(size_t)n * 16 + (j - 16)] = sigmoid_f(r);
}

// ---------------------------------------------------------------------------
// Node vector update: v_out = (v@Wsc_v + agg_v@Wlin_v)/4 * gate
// ---------------------------------------------------------------------------
__global__ void node_v_kernel(const float* __restrict__ v_in, const float* __restrict__ agg,
                              const float* __restrict__ Wsc_v, const float* __restrict__ Wlin_v,
                              const float* __restrict__ gate, float* __restrict__ v_out, int N)
{
    int tid = blockIdx.x * blockDim.x + threadIdx.x;
    if (tid >= N * 16) return;
    int n = tid >> 4, d = tid & 15;
    const float* vr = v_in + (size_t)n * 48;
    const float* ar = agg + (size_t)n * 80 + 32;
    float ox = 0.0f, oy = 0.0f, oz = 0.0f;
    #pragma unroll
    for (int c = 0; c < 16; ++c) {
        float ws = Wsc_v[c * 16 + d], wl = Wlin_v[c * 16 + d];
        ox += vr[c * 3 + 0] * ws + ar[c * 3 + 0] * wl;
        oy += vr[c * 3 + 1] * ws + ar[c * 3 + 1] * wl;
        oz += vr[c * 3 + 2] * ws + ar[c * 3 + 2] * wl;
    }
    float g = gate[(size_t)n * 16 + d] * 0.25f;
    float* vo = v_out + (size_t)n * 48 + d * 3;
    vo[0] = ox * g; vo[1] = oy * g; vo[2] = oz * g;
}

// ---------------------------------------------------------------------------
// Output head: (u, v, p)
// ---------------------------------------------------------------------------
__global__ void out_kernel(const float* __restrict__ aggF, const float* __restrict__ s,
                           const float* __restrict__ v, const float* __restrict__ Wsc_p,
                           const float* __restrict__ Wsc_vf, const float* __restrict__ w_vec,
                           const float* __restrict__ w_p, float* __restrict__ out, int N)
{
    int n = blockIdx.x * blockDim.x + threadIdx.x;
    if (n >= N) return;
    float p  = aggF[(size_t)n * 4 + 0];
    float ox = aggF[(size_t)n * 4 + 1];
    float oy = aggF[(size_t)n * 4 + 2];
    float sp = 0.0f, sx = 0.0f, sy = 0.0f;
    #pragma unroll
    for (int c = 0; c < 16; ++c) {
        sp += s[(size_t)n * 16 + c] * Wsc_p[c];
        float wc = Wsc_vf[c];
        sx += v[(size_t)n * 48 + c * 3 + 0] * wc;
        sy += v[(size_t)n * 48 + c * 3 + 1] * wc;
    }
    p += sp * 0.25f; ox += sx * 0.25f; oy += sy * 0.25f;
    float wvs = w_vec[0], wps = w_p[0];
    out[(size_t)n * 3 + 0] = ox * wvs;
    out[(size_t)n * 3 + 1] = oy * wvs;
    out[(size_t)n * 3 + 2] = p * wps;
}

// ---------------------------------------------------------------------------
// Host orchestration (all on `stream`; graph-capture safe: no malloc/memset API)
// Input order: x, embed, 3x{W1,W2,Wsc_s,Wsc_v,Wlin_s,Wlin_v},
//              final{W1,W2,Wsc_p,Wsc_v}, w_vec, w_p, src, dst
// ---------------------------------------------------------------------------
extern "C" void kernel_launch(void* const* d_in, const int* in_sizes, int n_in,
                              void* d_out, int out_size, void* d_ws, size_t ws_size,
                              hipStream_t stream)
{
    const float* x     = (const float*)d_in[0];
    const float* embed = (const float*)d_in[1];
    const float* Wl[3][6];
    for (int L = 0; L < 3; ++L)
        for (int k = 0; k < 6; ++k) Wl[L][k] = (const float*)d_in[2 + L * 6 + k];
    const float* fW1   = (const float*)d_in[20];
    const float* fW2   = (const float*)d_in[21];
    const float* fWp   = (const float*)d_in[22];
    const float* fWv   = (const float*)d_in[23];
    const float* w_vec = (const float*)d_in[24];
    const float* w_p   = (const float*)d_in[25];
    const int*   src   = (const int*)d_in[26];
    const int*   dst   = (const int*)d_in[27];
    const int N = in_sizes[0] / 2;
    const int E = in_sizes[26];
    float* out = (float*)d_out;

    // workspace carve-out (256B aligned)
    char* base = (char*)d_ws;
    size_t off = 0;
    auto walloc = [&](size_t bytes) -> char* {
        char* p = base + off;
        off = (off + bytes + 255) & ~(size_t)255;
        return p;
    };
    float* Y1   = (float*)walloc((size_t)E * 3 * 4);
    float* elen = (float*)walloc((size_t)E * 4);
    float* sA   = (float*)walloc((size_t)N * 16 * 4);
    float* sB   = (float*)walloc((size_t)N * 16 * 4);
    float* gate = (float*)walloc((size_t)N * 16 * 4);
    float* vA   = (float*)walloc((size_t)N * 48 * 4);
    float* vB   = (float*)walloc((size_t)N * 48 * 4);
    float* agg  = (float*)walloc((size_t)N * 80 * 4);
    _Float16* frB1[4];
    for (int i = 0; i < 4; ++i) frB1[i] = (_Float16*)walloc((size_t)7 * 32 * 16 * 2);
    _Float16* frB2[3];
    for (int i = 0; i < 3; ++i) frB2[i] = (_Float16*)walloc((size_t)4 * 6 * 32 * 16 * 2);
    _Float16* frB2F = (_Float16*)walloc((size_t)4 * 4 * 32 * 16 * 2);

    const dim3 blk(256);
    const float S1 = 0.31622777f;  // 1/sqrt(NB)
    const float S2 = 0.1f;         // 1/sqrt(HID)

    geom_kernel<<<dim3((E + 255) / 256), blk, 0, stream>>>(x, src, dst, Y1, elen, E);
    for (int L = 0; L < 3; ++L) {
        pack_b_kernel<<<dim3((7 * 512 + 255) / 256), blk, 0, stream>>>(Wl[L][0], frB1[L], 10, 100, 1, 7, S1);
        pack_b_kernel<<<dim3((24 * 512 + 255) / 256), blk, 0, stream>>>(Wl[L][1], frB2[L], 100, 96, 4, 6, S2);
    }
    pack_b_kernel<<<dim3((7 * 512 + 255) / 256), blk, 0, stream>>>(fW1, frB1[3], 10, 100, 1, 7, S1);
    pack_b_kernel<<<dim3((16 * 512 + 255) / 256), blk, 0, stream>>>(fW2, frB2F, 100, 64, 4, 4, S2);

    fill_s0_kernel<<<dim3((N * 16 + 255) / 256), blk, 0, stream>>>(sA, embed, N * 16);
    zero_kernel<<<dim3((unsigned)(((long)N * 48 + 255) / 256)), blk, 0, stream>>>(vA, (long)N * 48);

    float* sCur = sA; float* sNxt = sB;
    float* vCur = vA; float* vNxt = vB;
    const int eb = (E + 127) / 128;

    for (int L = 0; L < 3; ++L) {
        zero_kernel<<<dim3((unsigned)(((long)N * 80 + 255) / 256)), blk, 0, stream>>>(agg, (long)N * 80);
        edge_conv_kernel<<<dim3(eb), blk, 0, stream>>>(elen, Y1, src, dst, sCur, vCur,
                                                       frB1[L], frB2[L], agg, E);
        node_s_kernel<<<dim3((N * 32 + 255) / 256), blk, 0, stream>>>(sCur, agg, Wl[L][2], Wl[L][4],
                                                                      sNxt, gate, N);
        node_v_kernel<<<dim3((N * 16 + 255) / 256), blk, 0, stream>>>(vCur, agg, Wl[L][3], Wl[L][5],
                                                                      gate, vNxt, N);
        float* t;
        t = sCur; sCur = sNxt; sNxt = t;
        t = vCur; vCur = vNxt; vNxt = t;
    }

    zero_kernel<<<dim3((unsigned)(((long)N * 4 + 255) / 256)), blk, 0, stream>>>(agg, (long)N * 4);
    edge_final_kernel<<<dim3(eb), blk, 0, stream>>>(elen, Y1, src, dst, sCur, vCur,
                                                    frB1[3], frB2F, agg, E);
    out_kernel<<<dim3((N + 255) / 256), blk, 0, stream>>>(agg, sCur, vCur, fWp, fWv, w_vec, w_p, out, N);

    (void)n_in; (void)out_size; (void)ws_size;
}
